// TMP_SMorphLayer_4037269258262
// MI455X (gfx1250) — matrix operand order repaired and checked by reading, not compiled, not run on hardware
//
#include <hip/hip_runtime.h>

typedef __attribute__((ext_vector_type(16))) _Float16 v16h;
typedef __attribute__((ext_vector_type(2)))  _Float16 v2h;
typedef __attribute__((ext_vector_type(8)))  float    v8f;
typedef __attribute__((ext_vector_type(4)))  float    v4f;

#define KH_    3
#define KW_    3
#define C_IN_  3
#define F_OUT_ 16
#define B_     8
#define H_     128
#define W_     128
#define HO_    126
#define WO_    126
#define P_     27              // C*KH*KW
#define TPR_   8               // tiles per output row (last overlaps: wo0=110)
#define NROWS_ (B_*HO_)        // 1008
#define NT2_   (NROWS_*TPR_)   // 8064 tiles of 16 in-row pixels
#define FSTRIDE_ 15876         // HO_*WO_ (feature stride in out)

// Workspace layout (bytes):
//   Bf : 8 frags * 32 lanes * 16 half = 8192 B     @ 0
//   Ad : NT2_*2048 B = 16,515,072 B                @ 8192
//   Ae : NT2_*2048 B                               @ 8192 + 16,515,072
#define AD_OFF_ 8192ull
#define AE_OFF_ (8192ull + (unsigned long long)NT2_ * 2048ull)

// ---------------------------------------------------------------------------
// Kernel 1: B matrices in fragment-linear layout.
// frag = m*4 + ks*2 + nt ; lane holds column nt*16+col, K rows ks*32+lh*16+i.
// Augmented contraction: rows q<27 multiply the x*e^x part of A (coef e^k for
// s1); rows 27..53 multiply the e^x part (k*e^k for s1, e^k for s0); 54..63 pad.
// ---------------------------------------------------------------------------
__global__ void smorph_prep_b(const float* __restrict__ k1,
                              const float* __restrict__ k2,
                              v16h* __restrict__ Bf) {
    int tid  = threadIdx.x;          // 0..255
    int lane = tid & 31;
    int frag = tid >> 5;             // 0..7
    int m  = frag >> 2;
    int ks = (frag >> 1) & 1;
    int nt = frag & 1;
    int col = lane & 15;
    int lh  = lane >> 4;
    const float* km = m ? k2 : k1;
    v16h f;
    #pragma unroll
    for (int i = 0; i < 16; ++i) {
        int q = ks * 32 + lh * 16 + i;
        float v = 0.f;
        if (q < 2 * P_) {
            int p  = (q < P_) ? q : q - P_;
            int c  = p / 9;
            int r  = p % 9;
            int kh = r / 3, kw = r % 3;
            // reference: k1f[p=(c,kh,kw)][f] = k1[kh][kw][c][f]
            float kv = km[((kh * KW_ + kw) * C_IN_ + c) * F_OUT_ + col];
            float e  = __expf(kv);
            if (nt == 0) v = (q < P_) ? e : kv * e;   // s1 column
            else         v = (q < P_) ? 0.f : e;      // s0 column
        }
        f[i] = (_Float16)v;
    }
    Bf[frag * 32 + lane] = f;
}

// ---------------------------------------------------------------------------
// Kernel 2: fused im2col + exp factorization in WMMA-fragment-linear fp16
// layout (in-row tiling).  One thread per packed dword (2 halfs), both morphs.
// Element index: ((tile2*2 + ks)*32 + lane)*16 + i
// ---------------------------------------------------------------------------
__global__ void __launch_bounds__(256)
smorph_stage_a(const float* __restrict__ x,
               v2h* __restrict__ Ad, v2h* __restrict__ Ae) {
    int t = blockIdx.x * blockDim.x + threadIdx.x;   // dword index
    if (t >= NT2_ * 512) return;                     // NT2_*1024 halfs / 2
    int chunk = t >> 3;                              // 16-half chunk
    int i0    = (t & 7) * 2;
    int lane  = chunk & 31;
    int tk    = chunk >> 5;
    int ks    = tk & 1;
    int tile2 = tk >> 1;
    int t8    = tile2 & (TPR_ - 1);
    int rowid = tile2 >> 3;                          // b*HO_ + ho
    int ho = rowid % HO_;
    int b  = rowid / HO_;
    int wo0 = (t8 == TPR_ - 1) ? (WO_ - 16) : (t8 * 16);
    int row = lane & 15;
    int lh  = lane >> 4;
    int wo  = wo0 + row;

    float vd[2], ve[2];
    #pragma unroll
    for (int j = 0; j < 2; ++j) {
        int i  = i0 + j;
        int kk = 16 * lh + i;
        int q  = ks * 32 + kk;
        float fd = 0.f, fe = 0.f;
        if (q < 2 * P_) {
            int p  = (q < P_) ? q : q - P_;
            int c  = p / 9;
            int r  = p % 9;
            int kh = r / 3, kw = r - kh * 3;
            float xv = x[((b * C_IN_ + c) * H_ + (ho + kh)) * W_ + (wo + kw)];
            float ep = __expf(xv);
            float em = __expf(-xv);
            if (q < P_) { fd = xv * ep;  fe = -xv * em; }  // (patch)*E rows
            else        { fd = ep;       fe = em;       }  // E rows
        }
        vd[j] = fd; ve[j] = fe;
    }
    v2h pd = { (_Float16)vd[0], (_Float16)vd[1] };
    v2h pe = { (_Float16)ve[0], (_Float16)ve[1] };
    Ad[t] = pd;
    Ae[t] = pe;
}

// ---------------------------------------------------------------------------
// Kernel 3: grid (HO_, B_), 8 waves/block, one in-row 16-pixel tile per wave.
// Straight-line: 8 B-frag + 4 A-frag b128 loads, 8 WMMAs, rcp+fma epilogue,
// two b128 stores.  No division, no branches, EXEC stays all-ones.
// ---------------------------------------------------------------------------
__global__ void __launch_bounds__(256)
smorph_main(const v16h* __restrict__ Bf,
            const v16h* __restrict__ Ad, const v16h* __restrict__ Ae,
            const float* __restrict__ bias, float* __restrict__ out) {
    const int lane = threadIdx.x & 31;
    const int t8   = threadIdx.x >> 5;     // tile within row
    const int ho   = blockIdx.x;
    const int b    = blockIdx.y;
    const int col  = lane & 15;
    const int lh   = lane >> 4;
    const int wo0  = (t8 == TPR_ - 1) ? (WO_ - 16) : (t8 * 16);
    const int tile2 = (b * HO_ + ho) * TPR_ + t8;

    // B fragments resident in VGPRs (frag = m*4 + ks*2 + nt)
    v16h bfrag[8];
    #pragma unroll
    for (int f = 0; f < 8; ++f) bfrag[f] = Bf[f * 32 + lane];
    const float bv = bias[col];

    v16h ad0 = Ad[(tile2 * 2 + 0) * 32 + lane];
    v16h ad1 = Ad[(tile2 * 2 + 1) * 32 + lane];
    v16h ae0 = Ae[(tile2 * 2 + 0) * 32 + lane];
    v16h ae1 = Ae[(tile2 * 2 + 1) * 32 + lane];

    v8f acc_d1 = {}, acc_d0 = {}, acc_e1 = {}, acc_e0 = {};
    acc_d1 = __builtin_amdgcn_wmma_f32_16x16x32_f16(false, ad0, false, bfrag[0], (short)0, acc_d1, false, false);
    acc_d0 = __builtin_amdgcn_wmma_f32_16x16x32_f16(false, ad0, false, bfrag[1], (short)0, acc_d0, false, false);
    acc_d1 = __builtin_amdgcn_wmma_f32_16x16x32_f16(false, ad1, false, bfrag[2], (short)0, acc_d1, false, false);
    acc_d0 = __builtin_amdgcn_wmma_f32_16x16x32_f16(false, ad1, false, bfrag[3], (short)0, acc_d0, false, false);
    acc_e1 = __builtin_amdgcn_wmma_f32_16x16x32_f16(false, ae0, false, bfrag[4], (short)0, acc_e1, false, false);
    acc_e0 = __builtin_amdgcn_wmma_f32_16x16x32_f16(false, ae0, false, bfrag[5], (short)0, acc_e0, false, false);
    acc_e1 = __builtin_amdgcn_wmma_f32_16x16x32_f16(false, ae1, false, bfrag[6], (short)0, acc_e1, false, false);
    acc_e0 = __builtin_amdgcn_wmma_f32_16x16x32_f16(false, ae1, false, bfrag[7], (short)0, acc_e0, false, false);

    // D layout: lane holds feature column f=col, pixel rows m = i + 8*lh;
    // all 16 pixels sit in one output row -> contiguous stores.
    float v[8];
    #pragma unroll
    for (int i = 0; i < 8; ++i) {
        float r0 = __builtin_amdgcn_rcpf(acc_d0[i]);
        float r1 = __builtin_amdgcn_rcpf(acc_e0[i]);
        v[i] = acc_d1[i] * r0 + acc_e1[i] * r1 + bv;
    }
    const int obase = (b * F_OUT_ + col) * FSTRIDE_ + ho * WO_ + wo0 + 8 * lh;
    v4f o0 = { v[0], v[1], v[2], v[3] };
    v4f o1 = { v[4], v[5], v[6], v[7] };
    *(v4f*)(out + obase)     = o0;
    *(v4f*)(out + obase + 4) = o1;
}

// ---------------------------------------------------------------------------
extern "C" void kernel_launch(void* const* d_in, const int* in_sizes, int n_in,
                              void* d_out, int out_size, void* d_ws, size_t ws_size,
                              hipStream_t stream) {
    const float* x    = (const float*)d_in[0];
    const float* k1   = (const float*)d_in[1];
    const float* k2   = (const float*)d_in[2];
    const float* bias = (const float*)d_in[3];

    char* ws = (char*)d_ws;
    v16h* Bf  = (v16h*)ws;
    v2h*  Ad2 = (v2h*)(ws + AD_OFF_);
    v2h*  Ae2 = (v2h*)(ws + AE_OFF_);

    smorph_prep_b<<<1, 256, 0, stream>>>(k1, k2, Bf);

    int stageDwords = NT2_ * 512;
    smorph_stage_a<<<(stageDwords + 255) / 256, 256, 0, stream>>>(x, Ad2, Ae2);

    // grid (HO_, B_), 8 waves per block, one in-row tile per wave
    smorph_main<<<dim3(HO_, B_), 256, 0, stream>>>(
        Bf, (const v16h*)Ad2, (const v16h*)Ae2, bias, (float*)d_out);
}
